// Aligner_37031208026338
// MI455X (gfx1250) — compile-verified, matrix operand order checked by reading, not run on hardware
//
#include <hip/hip_runtime.h>
#include <hip/hip_bf16.h>

// ---------------------------------------------------------------------------
// Problem constants (match the reference)
// ---------------------------------------------------------------------------
#define I_DIM   512           // encoder dim
#define H_DIM   256           // query dim
#define M_DIM   80            // frame dim
#define C_DIM   1024          // GRU hidden
#define C2_DIM  512           // C/2
#define LOC_DIM 32
#define KW      31
#define B_DIM   32
#define S_DIM   256
#define T_DIM   800
#define G3      (3 * C_DIM)   // 3072 (r,z,n gates)
#define XK      (H_DIM + I_DIM)         // 768  GRU input K
#define TK_RAW  (I_DIM + M_DIM + C_DIM) // 1616
#define TK      1632          // padded to multiple of 32 for WMMA K

typedef __bf16  v16bf __attribute__((ext_vector_type(16)));
typedef float   v8f   __attribute__((ext_vector_type(8)));

union FragU { v16bf v; unsigned int u[8]; };

// ---------------------------------------------------------------------------
// bf16 WMMA GEMM:  C[M,N] = A[M,K] * W[N,K]^T (+ bias[N])
// A, W row-major bf16, leading dim == K.  K % 32 == 0, M % 32 == 0, N % 16 == 0.
// Each wave owns ONE 16-wide N strip and TWO M-tiles (M-tile pair), sharing a
// single B fragment per K-step across two WMMAs -> W (the dominant L2 traffic)
// is read exactly once per GEMM.  8 waves / 256-thread block.
// Fragment layouts follow the CDNA5 ISA 16-bit A 16x32 / f32 C 16x16 tables.
// ---------------------------------------------------------------------------
__global__ void gemm_bf16_kernel(const __hip_bfloat16* __restrict__ A,
                                 const __hip_bfloat16* __restrict__ W,
                                 const float* __restrict__ bias,
                                 float* __restrict__ C,
                                 int M, int N, int K) {
    const int lane = threadIdx.x & 31;
    const int wave = blockIdx.x * (blockDim.x >> 5) + (threadIdx.x >> 5);
    const int ntn  = N >> 4;          // N tiles
    const int ntm2 = M >> 5;          // M tile-pairs
    if (wave >= ntm2 * ntn) return;
    const int tm   = (wave / ntn) << 1;  // first M-tile of the pair
    const int tn   = wave % ntn;
    const int sub  = lane & 15;          // M index (A) / N index (B) within tile
    const int half = lane >> 4;          // lane-half selects K sub-range
    const int koff = half << 3;          // +8 K offset for lanes 16..31

    const __hip_bfloat16* arow0 = A + (size_t)(tm * 16 + sub) * K;
    const __hip_bfloat16* arow1 = arow0 + (size_t)16 * K;
    const __hip_bfloat16* wrow  = W + (size_t)(tn * 16 + sub) * K;

    v8f acc0 = {0.f, 0.f, 0.f, 0.f, 0.f, 0.f, 0.f, 0.f};
    v8f acc1 = {0.f, 0.f, 0.f, 0.f, 0.f, 0.f, 0.f, 0.f};

    for (int k0 = 0; k0 < K; k0 += 32) {
        FragU a0, a1, b;
#pragma unroll
        for (int v = 0; v < 8; ++v) {
            // VGPR v holds K = {base, base+1}; base = [0..7 | 16..23] + koff
            const int k = k0 + koff + ((v & 4) ? 16 : 0) + ((v & 3) << 1);
            a0.u[v] = *reinterpret_cast<const unsigned int*>(arow0 + k);
            a1.u[v] = *reinterpret_cast<const unsigned int*>(arow1 + k);
            b.u[v]  = *reinterpret_cast<const unsigned int*>(wrow + k);
        }
        acc0 = __builtin_amdgcn_wmma_f32_16x16x32_bf16(
            false, a0.v, false, b.v, (short)0, acc0, false, false);
        acc1 = __builtin_amdgcn_wmma_f32_16x16x32_bf16(
            false, a1.v, false, b.v, (short)0, acc1, false, false);
    }

    const int col = tn * 16 + sub;
    const float bv = bias ? bias[col] : 0.f;
#pragma unroll
    for (int r = 0; r < 8; ++r) {
        const int row = tm * 16 + r + (half << 3);   // C layout: M = r + 8*half
        C[(size_t)row * N + col]        = acc0[r] + bv;
        C[(size_t)(row + 16) * N + col] = acc1[r] + bv;
    }
}

// ---------------------------------------------------------------------------
// Setup kernels
// ---------------------------------------------------------------------------
__global__ void f32_to_bf16_kernel(const float* __restrict__ src,
                                   __hip_bfloat16* __restrict__ dst, int n) {
    const int i = blockIdx.x * blockDim.x + threadIdx.x;
    if (i < n) dst[i] = __float2bfloat16(src[i]);
}

// w_t1 [C, 1616] -> bf16 [C, 1632] with zero K-padding
__global__ void pad_wt1_kernel(const float* __restrict__ src,
                               __hip_bfloat16* __restrict__ dst) {
    const int i = blockIdx.x * blockDim.x + threadIdx.x;
    if (i >= C_DIM * TK) return;
    const int r = i / TK, c = i % TK;
    const float v = (c < TK_RAW) ? src[r * TK_RAW + c] : 0.f;
    dst[i] = __float2bfloat16(v);
}

__global__ void init_state_kernel(float* __restrict__ h,
                                  float* __restrict__ align,
                                  float* __restrict__ alpha,
                                  float* __restrict__ trans) {
    const int i = blockIdx.x * blockDim.x + threadIdx.x;
    if (i < B_DIM * C_DIM) h[i] = 0.f;
    if (i < B_DIM * S_DIM) {
        align[i] = 1.f / (float)S_DIM;
        alpha[i] = ((i & (S_DIM - 1)) == 0) ? 1.f : 0.f;
    }
    if (i < B_DIM) trans[i] = 0.5f;
}

// ---------------------------------------------------------------------------
// Per-step kernels
// ---------------------------------------------------------------------------
// prev = enc^T @ alpha; build xcat=[query_t | prev] (bf16) and hbf=bf16(h)
__global__ void pre_kernel(const float* __restrict__ enc,
                           const float* __restrict__ queries,
                           const float* __restrict__ h,
                           const float* __restrict__ alpha,
                           __hip_bfloat16* __restrict__ xcat,
                           __hip_bfloat16* __restrict__ hbf,
                           int t) {
    __shared__ float al[S_DIM];
    const int b = blockIdx.x, tid = threadIdx.x;
    al[tid] = alpha[b * S_DIM + tid];
    __syncthreads();
    float acc0 = 0.f, acc1 = 0.f;
    const float* eb = enc + (size_t)b * S_DIM * I_DIM;
    for (int s = 0; s < S_DIM; ++s) {
        const float a = al[s];
        acc0 += eb[(size_t)s * I_DIM + tid]       * a;
        acc1 += eb[(size_t)s * I_DIM + tid + 256] * a;
    }
    xcat[b * XK + tid] =
        __float2bfloat16(queries[((size_t)b * T_DIM + t) * H_DIM + tid]);
    xcat[b * XK + H_DIM + tid]       = __float2bfloat16(acc0);
    xcat[b * XK + H_DIM + tid + 256] = __float2bfloat16(acc1);
    for (int c = tid; c < C_DIM; c += 256)
        hbf[b * C_DIM + c] = __float2bfloat16(h[b * C_DIM + c]);
}

// GRU gate nonlinearity: r,z,n -> h_new (fp32 + bf16)
__global__ void gru_gate_kernel(const float* __restrict__ gx,
                                const float* __restrict__ gh,
                                float* __restrict__ h,
                                __hip_bfloat16* __restrict__ hbf) {
    const int idx = blockIdx.x * blockDim.x + threadIdx.x;  // B*C = 32768
    const int b = idx >> 10, c = idx & (C_DIM - 1);
    const float xr = gx[b * G3 + c];
    const float xz = gx[b * G3 + C_DIM + c];
    const float xn = gx[b * G3 + 2 * C_DIM + c];
    const float hr = gh[b * G3 + c];
    const float hz = gh[b * G3 + C_DIM + c];
    const float hn = gh[b * G3 + 2 * C_DIM + c];
    const float r = 1.f / (1.f + __expf(-(xr + hr)));
    const float z = 1.f / (1.f + __expf(-(xz + hz)));
    const float n = tanhf(xn + r * hn);
    const float hnew = (1.f - z) * n + z * h[idx];
    h[idx]   = hnew;
    hbf[idx] = __float2bfloat16(hnew);
}

// cv[b,l,s] = conv1d(align, w_loc1, pad=15);  one block per (b,l)
__global__ void loc_conv_kernel(const float* __restrict__ align,
                                const float* __restrict__ w_loc1,
                                float* __restrict__ cv) {
    __shared__ float alh[S_DIM + KW - 1];
    __shared__ float wl[KW];
    const int b = blockIdx.x >> 5, l = blockIdx.x & 31;
    const int tid = threadIdx.x;
    for (int j = tid; j < S_DIM + KW - 1; j += 256) {
        const int s = j - (KW / 2);
        alh[j] = (s >= 0 && s < S_DIM) ? align[b * S_DIM + s] : 0.f;
    }
    if (tid < KW) wl[tid] = w_loc1[l * KW + tid];
    __syncthreads();
    float acc = 0.f;
#pragma unroll
    for (int k = 0; k < KW; ++k) acc += wl[k] * alh[tid + k];
    cv[((size_t)b * LOC_DIM + l) * S_DIM + tid] = acc;
}

// energy[b,s] = sum_c w_agg[c]*tanh(loc + q + key); one wave per (b,s)
__global__ void energy_kernel(const float* __restrict__ cv,
                              const float* __restrict__ qproj,
                              const float* __restrict__ key,
                              const float* __restrict__ w_loc2,
                              const float* __restrict__ w_agg,
                              const float* __restrict__ mask,
                              float* __restrict__ energy) {
    __shared__ float cvl[8][LOC_DIM];
    const int lane = threadIdx.x & 31;
    const int wib  = threadIdx.x >> 5;
    const int w    = blockIdx.x * 8 + wib;
    const int b    = w >> 8;
    const int s    = w & (S_DIM - 1);
    cvl[wib][lane] = cv[((size_t)b * LOC_DIM + lane) * S_DIM + s];
    __syncthreads();
    float acc = 0.f;
    for (int c = lane; c < C2_DIM; c += 32) {
        float loc = 0.f;
        const float* wr = w_loc2 + c * LOC_DIM;
#pragma unroll
        for (int l = 0; l < LOC_DIM; ++l) loc += wr[l] * cvl[wib][l];
        const float sc = loc + qproj[b * C2_DIM + c] +
                         key[((size_t)b * S_DIM + s) * C2_DIM + c];
        acc += w_agg[c] * tanhf(sc);
    }
#pragma unroll
    for (int off = 16; off >= 1; off >>= 1) acc += __shfl_xor(acc, off, 32);
    if (lane == 0)
        energy[b * S_DIM + s] = (mask[b * S_DIM + s] != 0.f) ? acc : -3.0e38f;
}

// softmax over S, alpha recursion, normalize, emit alphas[:,t,:]
__global__ void softmax_alpha_kernel(const float* __restrict__ energy,
                                     const float* __restrict__ trans,
                                     float* __restrict__ alpha,
                                     float* __restrict__ align,
                                     float* __restrict__ alphas_out,
                                     int t) {
    __shared__ float red[S_DIM];
    const int b = blockIdx.x, s = threadIdx.x;
    const float e = energy[b * S_DIM + s];
    red[s] = e; __syncthreads();
    for (int off = 128; off >= 1; off >>= 1) {
        if (s < off) red[s] = fmaxf(red[s], red[s + off]);
        __syncthreads();
    }
    const float mx = red[0]; __syncthreads();
    const float p = __expf(e - mx);
    red[s] = p; __syncthreads();
    for (int off = 128; off >= 1; off >>= 1) {
        if (s < off) red[s] += red[s + off];
        __syncthreads();
    }
    const float alg = p / red[0]; __syncthreads();
    const float tr  = trans[b];
    const float a_c = alpha[b * S_DIM + s];
    const float a_p = (s > 0) ? alpha[b * S_DIM + s - 1] : 0.f;
    float an = ((1.f - tr) * a_c + tr * a_p + 1e-7f) * alg;
    red[s] = an; __syncthreads();
    for (int off = 128; off >= 1; off >>= 1) {
        if (s < off) red[s] += red[s + off];
        __syncthreads();
    }
    an = an / red[0];
    __syncthreads();   // all alpha reads done before overwrite
    alpha[b * S_DIM + s] = an;
    align[b * S_DIM + s] = alg;
    alphas_out[((size_t)b * T_DIM + t) * S_DIM + s] = an;
}

// attend = enc^T @ alpha_new; build t_in = [attend|frame|h_new|pad] bf16
__global__ void attend_tin_kernel(const float* __restrict__ enc,
                                  const float* __restrict__ outputs,
                                  const float* __restrict__ h,
                                  const float* __restrict__ alpha,
                                  __hip_bfloat16* __restrict__ tin,
                                  int t) {
    __shared__ float al[S_DIM];
    __shared__ float att[I_DIM];
    const int b = blockIdx.x, tid = threadIdx.x;
    al[tid] = alpha[b * S_DIM + tid];
    __syncthreads();
    float acc0 = 0.f, acc1 = 0.f;
    const float* eb = enc + (size_t)b * S_DIM * I_DIM;
    for (int s = 0; s < S_DIM; ++s) {
        const float a = al[s];
        acc0 += eb[(size_t)s * I_DIM + tid]       * a;
        acc1 += eb[(size_t)s * I_DIM + tid + 256] * a;
    }
    att[tid] = acc0; att[tid + 256] = acc1;
    __syncthreads();
    for (int j = tid; j < TK; j += 256) {
        float v;
        if (j < I_DIM)               v = att[j];
        else if (j < I_DIM + M_DIM)  v = outputs[((size_t)b * T_DIM + t) * M_DIM + (j - I_DIM)];
        else if (j < TK_RAW)         v = h[b * C_DIM + (j - I_DIM - M_DIM)];
        else                         v = 0.f;
        tin[b * TK + j] = __float2bfloat16(v);
    }
}

// trans[b] = sigmoid( sum_c tanh(tt[b,c]) * w_t2[c] )
__global__ void trans_kernel(const float* __restrict__ tt,
                             const float* __restrict__ w_t2,
                             float* __restrict__ trans) {
    __shared__ float red[256];
    const int b = blockIdx.x, tid = threadIdx.x;
    float acc = 0.f;
    for (int c = tid; c < C_DIM; c += 256) acc += tanhf(tt[b * C_DIM + c]) * w_t2[c];
    red[tid] = acc; __syncthreads();
    for (int off = 128; off >= 1; off >>= 1) {
        if (tid < off) red[tid] += red[tid + off];
        __syncthreads();
    }
    if (tid == 0) trans[b] = 1.f / (1.f + __expf(-red[0]));
}

// ---------------------------------------------------------------------------
// Host driver
// ---------------------------------------------------------------------------
extern "C" void kernel_launch(void* const* d_in, const int* in_sizes, int n_in,
                              void* d_out, int out_size, void* d_ws, size_t ws_size,
                              hipStream_t stream) {
    (void)in_sizes; (void)n_in; (void)out_size; (void)ws_size;
    const float* enc     = (const float*)d_in[0];
    const float* queries = (const float*)d_in[1];
    const float* outputs = (const float*)d_in[2];
    const float* mask    = (const float*)d_in[3];
    const float* w_ih    = (const float*)d_in[4];
    const float* w_hh    = (const float*)d_in[5];
    const float* b_ih    = (const float*)d_in[6];
    const float* b_hh    = (const float*)d_in[7];
    const float* w_q     = (const float*)d_in[8];
    const float* w_loc1  = (const float*)d_in[9];
    const float* w_loc2  = (const float*)d_in[10];
    const float* w_k     = (const float*)d_in[11];
    const float* b_k     = (const float*)d_in[12];
    const float* w_agg   = (const float*)d_in[13];
    const float* w_t1    = (const float*)d_in[14];
    const float* b_t1    = (const float*)d_in[15];
    const float* w_t2    = (const float*)d_in[16];
    float* alphas        = (float*)d_out;

    // workspace carve-up (256B aligned)
    char* p = (char*)d_ws;
    auto take = [&](size_t bytes) -> char* {
        char* q = p;
        p += (bytes + 255) & ~(size_t)255;
        return q;
    };
    __hip_bfloat16* enc_bf  = (__hip_bfloat16*)take((size_t)B_DIM * S_DIM * I_DIM * 2);
    __hip_bfloat16* wih_bf  = (__hip_bfloat16*)take((size_t)G3 * XK * 2);
    __hip_bfloat16* whh_bf  = (__hip_bfloat16*)take((size_t)G3 * C_DIM * 2);
    __hip_bfloat16* wq_bf   = (__hip_bfloat16*)take((size_t)C2_DIM * C_DIM * 2);
    __hip_bfloat16* wk_bf   = (__hip_bfloat16*)take((size_t)C2_DIM * I_DIM * 2);
    __hip_bfloat16* wt1_bf  = (__hip_bfloat16*)take((size_t)C_DIM * TK * 2);
    float* key     = (float*)take((size_t)B_DIM * S_DIM * C2_DIM * 4);
    float* h       = (float*)take((size_t)B_DIM * C_DIM * 4);
    __hip_bfloat16* hbf  = (__hip_bfloat16*)take((size_t)B_DIM * C_DIM * 2);
    __hip_bfloat16* xcat = (__hip_bfloat16*)take((size_t)B_DIM * XK * 2);
    __hip_bfloat16* tin  = (__hip_bfloat16*)take((size_t)B_DIM * TK * 2);
    float* align_b = (float*)take((size_t)B_DIM * S_DIM * 4);
    float* alpha_b = (float*)take((size_t)B_DIM * S_DIM * 4);
    float* trans_b = (float*)take((size_t)B_DIM * 4);
    float* gx      = (float*)take((size_t)B_DIM * G3 * 4);
    float* gh      = (float*)take((size_t)B_DIM * G3 * 4);
    float* qproj   = (float*)take((size_t)B_DIM * C2_DIM * 4);
    float* cv      = (float*)take((size_t)B_DIM * LOC_DIM * S_DIM * 4);
    float* energy  = (float*)take((size_t)B_DIM * S_DIM * 4);
    float* tt      = (float*)take((size_t)B_DIM * C_DIM * 4);

    auto conv = [&](const float* s, __hip_bfloat16* d, int cnt) {
        f32_to_bf16_kernel<<<(cnt + 255) / 256, 256, 0, stream>>>(s, d, cnt);
    };
    auto gemm = [&](const __hip_bfloat16* A, const __hip_bfloat16* W,
                    const float* bias, float* C, int M, int N, int K) {
        const int tiles  = (M / 32) * (N / 16);   // one wave per M-tile pair
        const int blocks = (tiles + 7) / 8;
        gemm_bf16_kernel<<<blocks, 256, 0, stream>>>(A, W, bias, C, M, N, K);
    };

    // ---- one-time setup ----
    conv(enc,  enc_bf, B_DIM * S_DIM * I_DIM);
    conv(w_ih, wih_bf, G3 * XK);
    conv(w_hh, whh_bf, G3 * C_DIM);
    conv(w_q,  wq_bf,  C2_DIM * C_DIM);
    conv(w_k,  wk_bf,  C2_DIM * I_DIM);
    pad_wt1_kernel<<<(C_DIM * TK + 255) / 256, 256, 0, stream>>>(w_t1, wt1_bf);
    init_state_kernel<<<(B_DIM * C_DIM) / 256, 256, 0, stream>>>(h, align_b, alpha_b, trans_b);
    // key = enc @ w_k^T + b_k    (M=8192, N=512, K=512)
    gemm(enc_bf, wk_bf, b_k, key, B_DIM * S_DIM, C2_DIM, I_DIM);

    // ---- sequential scan ----
    for (int t = 0; t < T_DIM; ++t) {
        pre_kernel<<<B_DIM, 256, 0, stream>>>(enc, queries, h, alpha_b, xcat, hbf, t);
        gemm(xcat, wih_bf, b_ih, gx, B_DIM, G3, XK);       // gx
        gemm(hbf,  whh_bf, b_hh, gh, B_DIM, G3, C_DIM);    // gh
        gru_gate_kernel<<<(B_DIM * C_DIM) / 256, 256, 0, stream>>>(gx, gh, h, hbf);
        gemm(hbf, wq_bf, nullptr, qproj, B_DIM, C2_DIM, C_DIM);   // q = h_new @ w_q^T
        loc_conv_kernel<<<B_DIM * LOC_DIM, 256, 0, stream>>>(align_b, w_loc1, cv);
        energy_kernel<<<(B_DIM * S_DIM) / 8, 256, 0, stream>>>(
            cv, qproj, key, w_loc2, w_agg, mask, energy);
        softmax_alpha_kernel<<<B_DIM, 256, 0, stream>>>(
            energy, trans_b, alpha_b, align_b, alphas, t);
        attend_tin_kernel<<<B_DIM, 256, 0, stream>>>(enc, outputs, h, alpha_b, tin, t);
        gemm(tin, wt1_bf, b_t1, tt, B_DIM, C_DIM, TK);     // tanh-layer GEMM
        trans_kernel<<<B_DIM, 256, 0, stream>>>(tt, w_t2, trans_b);
    }
}
